// MultiHeadAlibiAttention_63986422775985
// MI455X (gfx1250) — matrix operand order repaired and checked by reading, not compiled, not run on hardware
//
#include <hip/hip_runtime.h>

typedef __attribute__((ext_vector_type(16))) __bf16 v16bf;
typedef __attribute__((ext_vector_type(8)))  float  v8f;

#define NHEAD  16
#define DH     64
#define SEQ    2048
#define DMODEL 1024
#define BM     128
#define NWAVE  8
#define CHUNK  64

// ---- helpers -------------------------------------------------------------

// Low 32 bits of a generic pointer to __shared__ = LDS byte offset.
__device__ __forceinline__ unsigned lds_addr(const void* p) {
  return (unsigned)(unsigned long long)p;
}

// ASYNCcnt-tracked DMA of 16 bytes global -> LDS, no VGPR data movement.
__device__ __forceinline__ void async_b128(unsigned lds, const void* g) {
  asm volatile("global_load_async_to_lds_b128 %0, %1, off"
               :: "v"(lds), "v"(g) : "memory");
}
__device__ __forceinline__ void wait_async0() {
  asm volatile("s_wait_asynccnt 0" ::: "memory");
}
__device__ __forceinline__ void wait_ds0() {
  asm volatile("s_wait_dscnt 0" ::: "memory");
}

// Butterfly reductions across each 16-lane half via v_permlane16_b32.
__device__ __forceinline__ float permrow(float x, unsigned lo, unsigned hi) {
  return __uint_as_float(__builtin_amdgcn_permlane16(
      __float_as_uint(x), __float_as_uint(x), lo, hi, false, false));
}
__device__ __forceinline__ float rowmax16(float x) {
  x = fmaxf(x, permrow(x, 0x67452301u, 0xEFCDAB89u));  // xor 1
  x = fmaxf(x, permrow(x, 0x54761032u, 0xDCFE98BAu));  // xor 2
  x = fmaxf(x, permrow(x, 0x32107654u, 0xBA98FEDCu));  // xor 4
  x = fmaxf(x, permrow(x, 0xFEDCBA98u, 0x76543210u));  // xor 8
  return x;
}
__device__ __forceinline__ float rowsum16(float x) {
  x += permrow(x, 0x67452301u, 0xEFCDAB89u);
  x += permrow(x, 0x54761032u, 0xDCFE98BAu);
  x += permrow(x, 0x32107654u, 0xBA98FEDCu);
  x += permrow(x, 0xFEDCBA98u, 0x76543210u);
  return x;
}

union FragU { v16bf v; unsigned u[8]; };
union H4   { __bf16 h[4]; uint2 u; };

// ---- preprocessing: f32 -> bf16 in HBM ------------------------------------

// Q * 0.125 -> bf16, K -> bf16 (same [b][t][d] layout)
__global__ __launch_bounds__(256)
void cvt_qk_kernel(const float* __restrict__ Qf, const float* __restrict__ Kf,
                   __bf16* __restrict__ Qb, __bf16* __restrict__ Kb, int n4) {
  const int i = blockIdx.x * blockDim.x + threadIdx.x;
  if (i >= n4) return;
  const float4 q = ((const float4*)Qf)[i];
  const float4 k = ((const float4*)Kf)[i];
  H4 qo, ko;
  qo.h[0] = (__bf16)(q.x * 0.125f); qo.h[1] = (__bf16)(q.y * 0.125f);
  qo.h[2] = (__bf16)(q.z * 0.125f); qo.h[3] = (__bf16)(q.w * 0.125f);
  ko.h[0] = (__bf16)k.x; ko.h[1] = (__bf16)k.y;
  ko.h[2] = (__bf16)k.z; ko.h[3] = (__bf16)k.w;
  ((uint2*)Qb)[i] = qo.u;
  ((uint2*)Kb)[i] = ko.u;
}

// V[b][s][h*64+d] -> Vt[b][h][d][s] bf16 (LDS-tiled transpose, coalesced both ways)
__global__ __launch_bounds__(256)
void transpose_v_kernel(const float* __restrict__ Vf, __bf16* __restrict__ Vt) {
  __shared__ __bf16 tile[64][66];     // pad: stride 33 dwords, conflict-free
  const int s0 = blockIdx.x * 64;
  const int h  = blockIdx.y;
  const int b  = blockIdx.z;
  #pragma unroll
  for (int i = 0; i < 16; ++i) {
    const int e = i * 256 + threadIdx.x;
    const int s = e >> 6, d = e & 63;
    tile[s][d] = (__bf16)Vf[((size_t)(b * SEQ + s0 + s)) * DMODEL + h * DH + d];
  }
  __syncthreads();
  #pragma unroll
  for (int i = 0; i < 16; ++i) {
    const int e = i * 256 + threadIdx.x;
    const int d = e >> 6, s = e & 63;
    Vt[((size_t)(b * NHEAD + h) * DH + d) * SEQ + s0 + s] = tile[s][d];
  }
}

// ---- attention kernel -----------------------------------------------------

__global__ __launch_bounds__(256, 1)
void fa_alibi_kernel(const __bf16* __restrict__ Qb,
                     const __bf16* __restrict__ Kb,
                     const __bf16* __restrict__ Vt,
                     float* __restrict__ O)
{
  const int qb   = blockIdx.x * BM;
  const int h    = blockIdx.y;
  const int b    = blockIdx.z;
  const int tid  = threadIdx.x;
  const int lane = tid & 31;
  const int wave = tid >> 5;
  const int hlf  = lane >> 4;
  const int ln   = lane & 15;

  // dynamic LDS: K [2][64][64] bf16 (16 KB) then V [2][64][64] bf16 (16 KB)
  extern __shared__ char dynsmem[];
  __bf16 (*Ksh)[CHUNK][DH] = (__bf16 (*)[CHUNK][DH])dynsmem;              // [s][d]
  __bf16 (*Vsh)[DH][CHUNK] = (__bf16 (*)[DH][CHUNK])(dynsmem + 2 * CHUNK * DH * 2); // [d][s]
  __shared__ __bf16 Ps[NWAVE][16][CHUNK];   // per-wave P transpose scratch

  const float slope = exp2f(-0.5f * (float)(h + 1));  // 2^(-8(h+1)/16)
  const int   qrow0 = qb + wave * 16;
  const __bf16* Kg0 = Kb + ((size_t)(b * SEQ)) * DMODEL + h * DH;
  const __bf16* Vg0 = Vt + ((size_t)(b * NHEAD + h) * DH) * SEQ;

  // hoisted affine pieces of the ALiBi bias: bias = (s - t)*slope
  float tSlope[8];
  #pragma unroll
  for (int r = 0; r < 8; ++r)
    tSlope[r] = (float)(qrow0 + r + 8 * hlf) * slope;
  float sAdd[4];
  #pragma unroll
  for (int st = 0; st < 4; ++st)
    sAdd[st] = (float)(st * 16 + ln) * slope;

  // ---- Q strip (16 x 64, pre-scaled bf16) -> two A-fragments ----
  FragU aq0, aq1;
  {
    const __bf16* qrow = Qb + ((size_t)(b * SEQ + qrow0 + ln)) * DMODEL + h * DH;
    #pragma unroll
    for (int i = 0; i < 8; ++i) {
      const int kd = ((i >= 4) ? 16 : 0) + hlf * 8 + (i & 3) * 2;
      aq0.u[i] = *(const unsigned*)(qrow + kd);
      aq1.u[i] = *(const unsigned*)(qrow + 32 + kd);
    }
  }

  v8f acc0 = {}, acc1 = {}, acc2 = {}, acc3 = {};
  float mrow[8], lrow[8];
  #pragma unroll
  for (int r = 0; r < 8; ++r) { mrow[r] = -1e30f; lrow[r] = 0.0f; }

  const int nchunks = (qb + BM) / CHUNK;   // causal: s <= qb+BM-1

  // ---- prefetch chunk 0 into buffer 0 (async DMA to LDS, bf16 tiles) ----
  #pragma unroll
  for (int i = 0; i < 2; ++i) {
    const int idx = i * 256 + tid;            // 16B units; 8 bf16 per unit
    const int row = idx >> 3;
    const int e8  = (idx & 7) * 8;
    async_b128(lds_addr(&Ksh[0][row][e8]), Kg0 + (size_t)row * DMODEL + e8);
    async_b128(lds_addr(&Vsh[0][row][e8]), Vg0 + (size_t)row * SEQ + e8);
  }

  for (int j = 0; j < nchunks; ++j) {
    const int s0  = j * CHUNK;
    const int buf = j & 1;

    wait_async0();        // this wave's DMA for chunk j done
    __syncthreads();      // all waves' DMA done; prev readers of buf done

    // issue async prefetch of chunk j+1 into the other buffer
    if (j + 1 < nchunks) {
      const __bf16* Kg = Kg0 + (size_t)(s0 + CHUNK) * DMODEL;
      const __bf16* Vg = Vg0 + (s0 + CHUNK);
      #pragma unroll
      for (int i = 0; i < 2; ++i) {
        const int idx = i * 256 + tid;
        const int row = idx >> 3;
        const int e8  = (idx & 7) * 8;
        async_b128(lds_addr(&Ksh[buf ^ 1][row][e8]), Kg + (size_t)row * DMODEL + e8);
        async_b128(lds_addr(&Vsh[buf ^ 1][row][e8]), Vg + (size_t)row * SEQ + e8);
      }
    }

    // ---- scores: four 16x16 tiles over this 64-key chunk ----
    float sc[4][8];
    const float s0slope = (float)s0 * slope;
    const bool  needMask = (s0 + CHUNK - 1) > qrow0;   // wave-uniform
    #pragma unroll
    for (int st = 0; st < 4; ++st) {
      FragU bk0, bk1;
      const int scol = st * 16 + ln;       // B-layout: lane = N (key)
      #pragma unroll
      for (int i = 0; i < 8; ++i) {
        const int d0 = ((i >= 4) ? 16 : 0) + hlf * 8 + (i & 3) * 2;   // B-layout K pair (head dim)
        bk0.u[i] = *(const unsigned*)&Ksh[buf][scol][d0];
        bk1.u[i] = *(const unsigned*)&Ksh[buf][scol][32 + d0];
      }
      v8f c = {};
      c = __builtin_amdgcn_wmma_f32_16x16x32_bf16(false, aq0.v, false, bk0.v, (short)0, c, false, false);
      c = __builtin_amdgcn_wmma_f32_16x16x32_bf16(false, aq1.v, false, bk1.v, (short)0, c, false, false);
      const float base = s0slope + sAdd[st];           // s*slope
      if (!needMask) {
        #pragma unroll
        for (int r = 0; r < 8; ++r)
          sc[st][r] = c[r] + (base - tSlope[r]);       // Q pre-scaled by 1/8
      } else {
        const int di = (s0 + st * 16 + ln) - qrow0 - 8 * hlf;  // s - t + r
        #pragma unroll
        for (int r = 0; r < 8; ++r) {
          const float bias = (di > r) ? -1e30f : (base - tSlope[r]);
          sc[st][r] = c[r] + bias;
        }
      }
    }

    // ---- online softmax (permlane16 butterflies) ----
    float scale[8];
    #pragma unroll
    for (int r = 0; r < 8; ++r) {
      const float mx = rowmax16(fmaxf(fmaxf(sc[0][r], sc[1][r]),
                                      fmaxf(sc[2][r], sc[3][r])));
      const float mnew  = fmaxf(mrow[r], mx);
      const float scold = __expf(mrow[r] - mnew);
      mrow[r]  = mnew;
      scale[r] = scold;
      const float p0 = __expf(sc[0][r] - mnew);
      const float p1 = __expf(sc[1][r] - mnew);
      const float p2 = __expf(sc[2][r] - mnew);
      const float p3 = __expf(sc[3][r] - mnew);
      sc[0][r] = p0; sc[1][r] = p1; sc[2][r] = p2; sc[3][r] = p3;
      lrow[r] = lrow[r] * scold + rowsum16((p0 + p1) + (p2 + p3));
    }
    #pragma unroll
    for (int r = 0; r < 8; ++r) {
      acc0[r] *= scale[r]; acc1[r] *= scale[r];
      acc2[r] *= scale[r]; acc3[r] *= scale[r];
    }

    // ---- P (C layout) -> per-wave LDS scratch -> two bf16 A-fragments ----
    #pragma unroll
    for (int st = 0; st < 4; ++st)
      #pragma unroll
      for (int r = 0; r < 8; ++r)
        Ps[wave][r + 8 * hlf][st * 16 + ln] = (__bf16)sc[st][r];
    wait_ds0();   // same-wave DS ops are in-order; compiler barrier included

    FragU ap0, ap1;
    #pragma unroll
    for (int i = 0; i < 8; ++i) {
      const int kb = ((i >= 4) ? 16 : 0) + hlf * 8 + (i & 3) * 2;
      ap0.u[i] = *(const unsigned*)&Ps[wave][ln][kb];
      ap1.u[i] = *(const unsigned*)&Ps[wave][ln][32 + kb];
    }

    // ---- PV: out(16x64) += P(16x64) @ V(64x64), 4 col tiles x 2 K-steps ----
    #pragma unroll
    for (int n = 0; n < 4; ++n) {
      FragU bva, bvb;
      #pragma unroll
      for (int i = 0; i < 8; ++i) {
        const int sp = ((i >= 4) ? 16 : 0) + hlf * 8 + (i & 3) * 2;   // B-layout K pair (key index)
        bva.u[i] = *(const unsigned*)&Vsh[buf][n * 16 + ln][sp];       // (s,s+1) contiguous
        bvb.u[i] = *(const unsigned*)&Vsh[buf][n * 16 + ln][sp + 32];
      }
      switch (n) {
        case 0:
          acc0 = __builtin_amdgcn_wmma_f32_16x16x32_bf16(false, ap0.v, false, bva.v, (short)0, acc0, false, false);
          acc0 = __builtin_amdgcn_wmma_f32_16x16x32_bf16(false, ap1.v, false, bvb.v, (short)0, acc0, false, false);
          break;
        case 1:
          acc1 = __builtin_amdgcn_wmma_f32_16x16x32_bf16(false, ap0.v, false, bva.v, (short)0, acc1, false, false);
          acc1 = __builtin_amdgcn_wmma_f32_16x16x32_bf16(false, ap1.v, false, bvb.v, (short)0, acc1, false, false);
          break;
        case 2:
          acc2 = __builtin_amdgcn_wmma_f32_16x16x32_bf16(false, ap0.v, false, bva.v, (short)0, acc2, false, false);
          acc2 = __builtin_amdgcn_wmma_f32_16x16x32_bf16(false, ap1.v, false, bvb.v, (short)0, acc2, false, false);
          break;
        case 3:
          acc3 = __builtin_amdgcn_wmma_f32_16x16x32_bf16(false, ap0.v, false, bva.v, (short)0, acc3, false, false);
          acc3 = __builtin_amdgcn_wmma_f32_16x16x32_bf16(false, ap1.v, false, bvb.v, (short)0, acc3, false, false);
          break;
      }
    }
  }

  // ---- epilogue: divide by row sum, store f32 ----
  #pragma unroll
  for (int r = 0; r < 8; ++r) {
    const float inv = 1.0f / lrow[r];
    const size_t base = ((size_t)(b * SEQ + qrow0 + r + 8 * hlf)) * DMODEL + h * DH + ln;
    O[base +  0] = acc0[r] * inv;
    O[base + 16] = acc1[r] * inv;
    O[base + 32] = acc2[r] * inv;
    O[base + 48] = acc3[r] * inv;
  }
}

// ---- launcher -------------------------------------------------------------

extern "C" void kernel_launch(void* const* d_in, const int* in_sizes, int n_in,
                              void* d_out, int out_size, void* d_ws, size_t ws_size,
                              hipStream_t stream) {
  (void)n_in; (void)out_size; (void)ws_size;
  const float* q = (const float*)d_in[0];
  const float* k = (const float*)d_in[1];
  const float* v = (const float*)d_in[2];
  float* out = (float*)d_out;
  const int B = in_sizes[0] / (SEQ * DMODEL);

  // workspace: Qb | Kb | Vt, each B*SEQ*DMODEL bf16 (24 MiB total for B=2)
  const size_t nElem = (size_t)B * SEQ * DMODEL;
  __bf16* Qb = (__bf16*)d_ws;
  __bf16* Kb = Qb + nElem;
  __bf16* Vt = Kb + nElem;

  // 1) Q*0.125 -> bf16, K -> bf16
  const int n4 = (int)(nElem / 4);
  cvt_qk_kernel<<<(n4 + 255) / 256, 256, 0, stream>>>(q, k, Qb, Kb, n4);
  // 2) V -> Vt[b][h][d][s] bf16
  dim3 tg(SEQ / 64, NHEAD, B);
  transpose_v_kernel<<<tg, 256, 0, stream>>>(v, Vt);
  // 3) flash attention
  dim3 grid(SEQ / BM, NHEAD, B);
  const size_t dyn_lds = (size_t)4 * CHUNK * DH * sizeof(__bf16);  // 32 KB K+V
  fa_alibi_kernel<<<grid, 256, dyn_lds, stream>>>(Qb, Kb, Vt, out);
}
